// Head_42898133353079
// MI455X (gfx1250) — matrix operand order repaired and checked by reading, not compiled, not run on hardware
//
#include <hip/hip_runtime.h>

// Batched causal attention head, fully fused per batch element.
// B=4096, T=64, C=128 (n_embd), Hd=32. One block per batch, 128 threads (4 waves).
// Matmuls via v_wmma_f32_16x16x32_f16 (f16 operands, f32 accumulate).

#define T_SEQ 64
#define C_EMB 128
#define HD    32

// LDS row strides (in elements), all multiples of 8 halfs -> 16B-aligned rows.
#define LDX  136   // x tile     [64][136] f16
#define LDW  136   // W^T tiles  [32][136] f16 (k-major per output column)
#define LDQ  40    // Q          [64][40]  f16
#define LDK  40    // K          [64][40]  f16
#define LDV  72    // V^T        [32][72]  f16  (Vt[d][s])
#define LDSF 68    // scores     [64][68]  f32
#define LDP  72    // probs      [64][72]  f16

typedef __attribute__((ext_vector_type(16))) _Float16 v16h;
typedef __attribute__((ext_vector_type(8)))  _Float16 v8h;
typedef __attribute__((ext_vector_type(8)))  float    v8f;

// Live-range overlap: x tile dies before scores are written; W^T tiles die
// before probabilities are written.
union SmemA { _Float16 Xh[T_SEQ * LDX]; float Sf[T_SEQ * LDSF]; };     // 17408 B
union SmemB { _Float16 Wt[3][HD * LDW]; _Float16 Ps[T_SEQ * LDP]; };   // 26112 B

// A-fragment (16x32 f16, MxK): lane l holds row m=l&15; half hi=l>>4 selects
// k in {8*hi .. 8*hi+7} (VGPR 0-3) and {16+8*hi .. 16+8*hi+7} (VGPR 4-7).
__device__ __forceinline__ v16h load_frag_a(const _Float16* base, int ld) {
  const int lane = threadIdx.x & 31;
  const int m  = lane & 15;
  const int hi = lane >> 4;
  const _Float16* row = base + m * ld + hi * 8;
  v8h lo = *(const v8h*)(row);        // k = 8*hi + 0..7
  v8h hv = *(const v8h*)(row + 16);   // k = 16 + 8*hi + 0..7
  v16h r;
#pragma unroll
  for (int i = 0; i < 8; ++i) { r[i] = lo[i]; r[i + 8] = hv[i]; }
  return r;
}

// B-fragment (32x16 f16, KxN) fetched from a transposed [N][K] LDS image:
// lane l holds column n=l&15, k = kbase + 8*(l>>4) + 0..7 and kbase + 16 + 8*(l>>4) + 0..7.
__device__ __forceinline__ v16h load_frag_bt(const _Float16* base, int ld, int kbase) {
  const int lane = threadIdx.x & 31;
  const int n  = lane & 15;
  const int hi = lane >> 4;
  const _Float16* p = base + n * ld + kbase + hi * 8;
  v8h lo = *(const v8h*)(p);
  v8h hv = *(const v8h*)(p + 16);
  v16h r;
#pragma unroll
  for (int i = 0; i < 8; ++i) { r[i] = lo[i]; r[i + 8] = hv[i]; }
  return r;
}

#define WMMA(A, B, C) \
  __builtin_amdgcn_wmma_f32_16x16x32_f16(false, (A), false, (B), (short)0, (C), false, false)

__global__ __launch_bounds__(128)
void attn_head_kernel(const float* __restrict__ x,
                      const float* __restrict__ Wq,
                      const float* __restrict__ Wk,
                      const float* __restrict__ Wv,
                      float* __restrict__ out) {
  __shared__ alignas(16) SmemA sA;
  __shared__ alignas(16) SmemB sB;
  __shared__ alignas(16) _Float16 Qs[T_SEQ * LDQ];
  __shared__ alignas(16) _Float16 Ks[T_SEQ * LDK];
  __shared__ alignas(16) _Float16 Vt[HD * LDV];

  const int tid  = threadIdx.x;
  const int wave = tid >> 5;
  const int lane = tid & 31;
  const int ln   = lane & 15;
  const int lh   = lane >> 4;
  const size_t b = blockIdx.x;
  const int m0   = wave * 16;

  // ---------------- Phase 1: stage x (f32->f16) and W^T into LDS ----------
  {
    const float4* xb = (const float4*)(x + b * (size_t)(T_SEQ * C_EMB));
#pragma unroll
    for (int i = 0; i < 16; ++i) {
      int idx = tid + i * 128;          // 2048 float4 per tile
      float4 f = xb[idx];
      int e   = idx * 4;
      int row = e >> 7;                 // /128
      int col = e & 127;
      _Float16* d = &sA.Xh[row * LDX + col];
      d[0] = (_Float16)f.x; d[1] = (_Float16)f.y;
      d[2] = (_Float16)f.z; d[3] = (_Float16)f.w;
    }
    // W: [128][32] f32, contiguous -> float4 loads (8 per thread per matrix),
    // scattered as f16 into the transposed [N][K] LDS image.
    const float4* Ws4[3] = {(const float4*)Wq, (const float4*)Wk, (const float4*)Wv};
#pragma unroll
    for (int p = 0; p < 3; ++p) {
#pragma unroll
      for (int i = 0; i < 8; ++i) {
        int idx = tid + i * 128;        // 1024 float4 of [128][32]
        float4 f = Ws4[p][idx];
        int e = idx * 4;
        int k = e >> 5;                 // row of W = k index
        int n = e & 31;                 // output-column base (multiple of 4)
        _Float16* col = &sB.Wt[p][n * LDW + k];
        col[0 * LDW] = (_Float16)f.x;
        col[1 * LDW] = (_Float16)f.y;
        col[2 * LDW] = (_Float16)f.z;
        col[3 * LDW] = (_Float16)f.w;
      }
    }
  }
  __syncthreads();

  // ---------------- Phase 2: Q,K,V = x @ W (wave owns rows m0..m0+15) -----
  {
    v8f aQ0 = {}, aQ1 = {}, aK0 = {}, aK1 = {}, aV0 = {}, aV1 = {};
#pragma unroll
    for (int kk = 0; kk < 4; ++kk) {                    // K = 128 in 4 steps
      v16h a = load_frag_a(&sA.Xh[m0 * LDX + kk * 32], LDX);
      v16h b0 = load_frag_bt(&sB.Wt[0][0],        LDW, kk * 32);
      v16h b1 = load_frag_bt(&sB.Wt[0][16 * LDW], LDW, kk * 32);
      aQ0 = WMMA(a, b0, aQ0);  aQ1 = WMMA(a, b1, aQ1);
      b0 = load_frag_bt(&sB.Wt[1][0],        LDW, kk * 32);
      b1 = load_frag_bt(&sB.Wt[1][16 * LDW], LDW, kk * 32);
      aK0 = WMMA(a, b0, aK0);  aK1 = WMMA(a, b1, aK1);
      b0 = load_frag_bt(&sB.Wt[2][0],        LDW, kk * 32);
      b1 = load_frag_bt(&sB.Wt[2][16 * LDW], LDW, kk * 32);
      aV0 = WMMA(a, b0, aV0);  aV1 = WMMA(a, b1, aV1);
    }
    // D layout: lane holds (m = r + 8*lh, n = ln) in VGPR r.
#pragma unroll
    for (int r = 0; r < 8; ++r) {
      int m = m0 + r + 8 * lh;
      Qs[m * LDQ + ln]        = (_Float16)aQ0[r];
      Qs[m * LDQ + 16 + ln]   = (_Float16)aQ1[r];
      Ks[m * LDK + ln]        = (_Float16)aK0[r];
      Ks[m * LDK + 16 + ln]   = (_Float16)aK1[r];
      Vt[ln * LDV + m]        = (_Float16)aV0[r];      // V^T: [d][s]
      Vt[(16 + ln) * LDV + m] = (_Float16)aV1[r];
    }
  }
  __syncthreads();

  // ---------------- Phase 3: S = Q @ K^T (raw scores, f32) ----------------
  {
    v16h a = load_frag_a(&Qs[m0 * LDQ], LDQ);           // 16x32, full Hd
#pragma unroll
    for (int st = 0; st < 4; ++st) {                    // 4 column tiles of s
      v16h bf = load_frag_bt(&Ks[st * 16 * LDK], LDK, 0);
      v8f s = {};
      s = WMMA(a, bf, s);
#pragma unroll
      for (int r = 0; r < 8; ++r)
        sA.Sf[(m0 + r + 8 * lh) * LDSF + st * 16 + ln] = s[r];
    }
  }
  __syncthreads();

  // ---------------- Phase 4: causal softmax per row (threads 0..63) -------
  if (tid < T_SEQ) {
    const int i = tid;
    const float scale = 0.17677669529663687f;           // 1/sqrt(32)
    float mx = -3.4e38f;
    for (int j = 0; j <= i; ++j)
      mx = fmaxf(mx, sA.Sf[i * LDSF + j] * scale);
    float sum = 0.f;
    for (int j = 0; j <= i; ++j)
      sum += __expf(sA.Sf[i * LDSF + j] * scale - mx);
    float inv = 1.0f / sum;
    for (int j = 0; j < T_SEQ; ++j) {
      float p = (j <= i) ? __expf(sA.Sf[i * LDSF + j] * scale - mx) * inv : 0.0f;
      sB.Ps[i * LDP + j] = (_Float16)p;                 // overwrites dead W^T
    }
  }
  __syncthreads();

  // ---------------- Phase 5: O = P @ V ------------------------------------
  {
    v8f o0 = {}, o1 = {};
#pragma unroll
    for (int kk = 0; kk < 2; ++kk) {                    // K = 64 in 2 steps
      v16h a  = load_frag_a(&sB.Ps[m0 * LDP + kk * 32], LDP);
      v16h b0 = load_frag_bt(&Vt[0],        LDV, kk * 32);
      v16h b1 = load_frag_bt(&Vt[16 * LDV], LDV, kk * 32);
      o0 = WMMA(a, b0, o0);
      o1 = WMMA(a, b1, o1);
    }
    float* ob = out + b * (size_t)(T_SEQ * HD);
#pragma unroll
    for (int r = 0; r < 8; ++r) {
      int m = m0 + r + 8 * lh;
      ob[m * HD + ln]      = o0[r];
      ob[m * HD + 16 + ln] = o1[r];
    }
  }
}

extern "C" void kernel_launch(void* const* d_in, const int* in_sizes, int n_in,
                              void* d_out, int out_size, void* d_ws, size_t ws_size,
                              hipStream_t stream) {
  const float* x  = (const float*)d_in[0];
  const float* Wq = (const float*)d_in[1];
  const float* Wk = (const float*)d_in[2];
  const float* Wv = (const float*)d_in[3];
  float* out = (float*)d_out;
  const int B = in_sizes[0] / (T_SEQ * C_EMB);          // 4096
  attn_head_kernel<<<dim3(B), dim3(128), 0, stream>>>(x, Wq, Wk, Wv, out);
}
